// PointConvTransFlowV3_9354438770931
// MI455X (gfx1250) — matrix operand order repaired and checked by reading, not compiled, not run on hardware
//
#include <hip/hip_runtime.h>

#define BATCH 2
#define NPTS  8192
#define DIMF  64
#define KNN   16
#define RTOT  (BATCH*NPTS*KNN)   // 262144 rows (b,n,k) with k fastest
#define R2    (BATCH*NPTS)       // 16384
#define WPB   4                  // waves per block in WMMA kernels

typedef __attribute__((ext_vector_type(16))) _Float16 v16h;
typedef __attribute__((ext_vector_type(8)))  _Float16 v8h;
typedef __attribute__((ext_vector_type(8)))  float    v8f;

#define WMMA(a,b,c) __builtin_amdgcn_wmma_f32_16x16x32_f16(false,(a),false,(b),(short)0,(c),false,false)

// ---- WMMA fragment loads (layouts per CDNA5 ISA 7.12.2) -------------------
// A (16x32 f16): lane<16 -> row=lane, K {kb..kb+7, kb+16..kb+23}
//                lane>=16 -> row=lane-16, K {kb+8..kb+15, kb+24..kb+31}
__device__ __forceinline__ v16h ldA(const _Float16* base, int ld, int kb, int lane) {
  int m  = lane & 15;
  int sh = (lane & 16) ? 8 : 0;
  const _Float16* p = base + (size_t)m * ld + kb + sh;
  v8h lo = *(const v8h*)(p);
  v8h hi = *(const v8h*)(p + 16);
  return __builtin_shufflevector(lo, hi, 0,1,2,3,4,5,6,7,8,9,10,11,12,13,14,15);
}
// B (32x16 f16) from pre-transposed weights WT[Cout][CinPad]:
// lane<16 -> col=lane, K {kb..kb+7, kb+16..kb+23} ; lane>=16 -> col=lane-16, K {kb+8..kb+15, kb+24..kb+31}
__device__ __forceinline__ v16h ldB(const _Float16* wt, int ld, int ncol, int kb, int lane) {
  int n  = lane & 15;
  int sh = (lane & 16) ? 8 : 0;
  const _Float16* p = wt + (size_t)(ncol + n) * ld + kb + sh;
  v8h lo = *(const v8h*)(p);
  v8h hi = *(const v8h*)(p + 16);
  return __builtin_shufflevector(lo, hi, 0,1,2,3,4,5,6,7,8,9,10,11,12,13,14,15);
}

// ---- weight prep: W[Cin][Cout] f32 -> WT[Cout][CinPad] f16 (zero padded) --
__global__ void prep_wT(const float* __restrict__ W, _Float16* __restrict__ WT,
                        int Cin, int Cout, int CinPad) {
  int t = blockIdx.x * 256 + threadIdx.x;
  if (t >= Cout * CinPad) return;
  int o = t / CinPad, k = t - o * CinPad;
  WT[(size_t)o * CinPad + k] = (k < Cin) ? (_Float16)W[(size_t)k * Cout + o] : (_Float16)0.0f;
}

// ---- brute force KNN, top-16 in sorted registers, LDS tiles of refs -------
__global__ void __launch_bounds__(256) knn_kernel(const float* __restrict__ q,
                                                  const float* __restrict__ sf,
                                                  const float* __restrict__ ref,
                                                  int* __restrict__ out) {
  __shared__ float sx[256], sy[256], sz[256];
  int b = blockIdx.y;
  int n = blockIdx.x * 256 + threadIdx.x;
  const float* qb = q   + (size_t)b * 3 * NPTS;
  const float* rb = ref + (size_t)b * 3 * NPTS;
  float q0 = qb[n], q1 = qb[NPTS + n], q2 = qb[2*NPTS + n];
  if (sf) {
    const float* sb = sf + (size_t)b * 3 * NPTS;
    q0 += sb[n]; q1 += sb[NPTS + n]; q2 += sb[2*NPTS + n];
  }
  float bd[KNN]; int bi[KNN];
  #pragma unroll
  for (int i = 0; i < KNN; i++) { bd[i] = 3.0e38f; bi[i] = 0; }
  for (int t = 0; t < NPTS; t += 256) {
    int j = t + threadIdx.x;
    sx[threadIdx.x] = rb[j];
    sy[threadIdx.x] = rb[NPTS + j];
    sz[threadIdx.x] = rb[2*NPTS + j];
    __syncthreads();
    for (int jj = 0; jj < 256; jj++) {
      float dx = q0 - sx[jj], dy = q1 - sy[jj], dz = q2 - sz[jj];
      float d = dx*dx + dy*dy + dz*dz;
      if (d < bd[KNN-1]) {
        float cd = d; int ci = t + jj;
        #pragma unroll
        for (int i = 0; i < KNN; i++) {
          if (cd < bd[i]) {
            float td = bd[i]; int ti = bi[i];
            bd[i] = cd; bi[i] = ci; cd = td; ci = ti;
          }
        }
      }
    }
    __syncthreads();
  }
  int* o = out + (size_t)(b * NPTS + n) * KNN;
  #pragma unroll
  for (int i = 0; i < KNN; i++) o[i] = bi[i];
}

// ---- gather direction vectors + seed padded ROWS3 -------------------------
__global__ void gather_dir(const float* __restrict__ x1, const float* __restrict__ x2,
                           const int* __restrict__ idx, const int* __restrict__ idxw,
                           float* __restrict__ dir, float* __restrict__ dirw,
                           _Float16* __restrict__ rows3) {
  int r = blockIdx.x * 256 + threadIdx.x;
  if (r >= RTOT) return;
  int b = r >> 17;                 // N*K = 2^17
  int n = (r >> 4) & (NPTS - 1);
  const float* x1b = x1 + (size_t)b * 3 * NPTS;
  const float* x2b = x2 + (size_t)b * 3 * NPTS;
  int j = idx[r], jw = idxw[r];
  #pragma unroll
  for (int c = 0; c < 3; c++) {
    float qv = x1b[c*NPTS + n];
    float d  = x2b[c*NPTS + j]  - qv;
    float dw = x2b[c*NPTS + jw] - qv;   // note: both gathered from x2 (per ref)
    dir [(size_t)r*3 + c] = d;
    dirw[(size_t)r*3 + c] = dw;
    rows3[(size_t)r*96 + c] = (_Float16)dw;
  }
  for (int c = 67; c < 96; c++) rows3[(size_t)r*96 + c] = (_Float16)0.0f;
}

// ---- gather concat(p1, p2[idx]) rows as f16 [R,128] -----------------------
__global__ void gather_feat(const float* __restrict__ p1, const float* __restrict__ p2,
                            const int* __restrict__ idx, _Float16* __restrict__ rows) {
  int r = blockIdx.x * 256 + threadIdx.x;
  if (r >= RTOT) return;
  int b = r >> 17;
  int n = (r >> 4) & (NPTS - 1);
  const float* p1b = p1 + (size_t)b * DIMF * NPTS;
  const float* p2b = p2 + (size_t)b * DIMF * NPTS;
  int j = idx[r];
  _Float16* row = rows + (size_t)r * 128;
  #pragma unroll 4
  for (int d = 0; d < DIMF; d++) {
    row[d]        = (_Float16)p1b[d * NPTS + n];
    row[DIMF + d] = (_Float16)p2b[d * NPTS + j];
  }
}

// ---- two-layer leaky MLP over 16-row tiles via WMMA -----------------------
__global__ void __launch_bounds__(128)
chain_kernel(const _Float16* __restrict__ X, int ldx,
             const _Float16* __restrict__ WaT, const float* __restrict__ ba,
             const _Float16* __restrict__ WbT, const float* __restrict__ bb,
             _Float16* __restrict__ outH, int ldo, int outOff,
             float* __restrict__ outFinal) {
  __shared__ __align__(32) _Float16 hbuf[WPB][16*64];
  int wave = threadIdx.x >> 5, lane = threadIdx.x & 31;
  size_t r0 = ((size_t)blockIdx.x * WPB + wave) * 16;
  const _Float16* Xb = X + r0 * ldx;
  _Float16* h = hbuf[wave];

  v8f acc[4] = {};
  for (int kc = 0; kc < ldx; kc += 32) {
    v16h a = ldA(Xb, ldx, kc, lane);
    #pragma unroll
    for (int nt = 0; nt < 4; nt++)
      acc[nt] = WMMA(a, ldB(WaT, ldx, nt*16, kc, lane), acc[nt]);
  }
  #pragma unroll
  for (int nt = 0; nt < 4; nt++) {
    int col = nt*16 + (lane & 15), rb = (lane & 16) ? 8 : 0;
    float bias = ba[col];
    #pragma unroll
    for (int v = 0; v < 8; v++) {
      float x = acc[nt][v] + bias;
      x = (x > 0.0f) ? x : 0.1f * x;
      h[(rb + v)*64 + col] = (_Float16)x;
    }
  }
  __syncthreads();
  v8f acc2[4] = {};
  #pragma unroll
  for (int kc = 0; kc < 64; kc += 32) {
    v16h a = ldA(h, 64, kc, lane);
    #pragma unroll
    for (int nt = 0; nt < 4; nt++)
      acc2[nt] = WMMA(a, ldB(WbT, 64, nt*16, kc, lane), acc2[nt]);
  }
  #pragma unroll
  for (int nt = 0; nt < 4; nt++) {
    int col = nt*16 + (lane & 15), rb = (lane & 16) ? 8 : 0;
    float bias = bb[col];
    #pragma unroll
    for (int v = 0; v < 8; v++) {
      float x = acc2[nt][v] + bias;
      x = (x > 0.0f) ? x : 0.1f * x;
      size_t r = r0 + rb + v;
      if (outH) outH[r*ldo + outOff + col] = (_Float16)x;
      if (outFinal) {                       // [B,64,N] transposed store
        size_t bb_ = r / NPTS, nn = r - bb_*NPTS;
        outFinal[(bb_*64 + col)*NPTS + nn] = x;
      }
    }
  }
}

// ---- IntraPatchCost: per (b,n) tile of 16 neighbors -----------------------
__global__ void __launch_bounds__(128)
ipc_kernel(const float* __restrict__ dir,
           const float* __restrict__ iw1, const float* __restrict__ ib1,
           const float* __restrict__ ig,  const float* __restrict__ ibt,
           const _Float16* __restrict__ ie2T, const float* __restrict__ ie2b,
           const _Float16* __restrict__ pecT, const float* __restrict__ pecb,
           const _Float16* __restrict__ pm1T, const float* __restrict__ pm1b,
           const float* __restrict__ pm1g, const float* __restrict__ pm1bt,
           const _Float16* __restrict__ pm2T, const float* __restrict__ pm2b,
           const float* __restrict__ pm2g, const float* __restrict__ pm2bt,
           const _Float16* __restrict__ pm3T, const float* __restrict__ pm3b,
           const float* __restrict__ pm3g, const float* __restrict__ pm3bt,
           const float* __restrict__ pm4w, const float* __restrict__ pm4b,
           const float* __restrict__ pm4g, const float* __restrict__ pm4bt,
           float* __restrict__ attn) {
  __shared__ __align__(32) float    pdl[WPB][48];
  __shared__ __align__(32) _Float16 xt [WPB][16*64];
  __shared__ __align__(32) _Float16 pef[WPB][16*32];
  __shared__ __align__(32) _Float16 h3s[WPB][16*32];
  __shared__ __align__(32) float    lg [WPB][16];
  int wave = threadIdx.x >> 5, lane = threadIdx.x & 31;
  size_t r0 = ((size_t)blockIdx.x * WPB + wave) * 16;

  for (int i = lane; i < 48; i += 32) pdl[wave][i] = dir[r0*3 + i];
  __syncthreads();

  // front-end 3->64 (BN eps 1e-5, relu) into xt
  const float s_ie = rsqrtf(1.0f + 1e-5f);
  for (int i = 0; i < 32; i++) {
    int e = lane + 32*i, row = e >> 6, col = e & 63;
    float v = pdl[wave][row*3+0]*iw1[col] + pdl[wave][row*3+1]*iw1[64+col]
            + pdl[wave][row*3+2]*iw1[128+col] + ib1[col];
    v = v * (ig[col]*s_ie) + ibt[col];
    xt[wave][row*64+col] = (_Float16)fmaxf(v, 0.0f);
  }
  // Fourier features 24 (pad 32) into pef: dim_t = 10000^{j/4} = {1,10,100,1000}
  const float dimt[4] = {1.0f, 10.0f, 100.0f, 1000.0f};
  for (int i = 0; i < 16; i++) {
    int e = lane + 32*i, row = e >> 5, j = e & 31;
    float f = 0.0f;
    if (j < 24) {
      int c = j >> 3, i8 = j & 7;
      float pv = pdl[wave][row*3 + c];
      float pe = (pv*4.0f - rintf(pv*4.0f)) * (6.283185307179586f / (1.0f + 1e-6f));
      float val = pe / dimt[i8 >> 1];
      f = (i8 & 1) ? cosf(val) : sinf(val);
    }
    pef[wave][row*32 + j] = (_Float16)f;
  }
  __syncthreads();

  // x = t1@ie_w2 + pef@pec_w + (ie_b2+pec_b)   (both GEMMs into same acc)
  v8f acc[4] = {};
  for (int kc = 0; kc < 64; kc += 32) {
    v16h a = ldA(xt[wave], 64, kc, lane);
    #pragma unroll
    for (int nt = 0; nt < 4; nt++)
      acc[nt] = WMMA(a, ldB(ie2T, 64, nt*16, kc, lane), acc[nt]);
  }
  {
    v16h a = ldA(pef[wave], 32, 0, lane);
    #pragma unroll
    for (int nt = 0; nt < 4; nt++)
      acc[nt] = WMMA(a, ldB(pecT, 32, nt*16, 0, lane), acc[nt]);
  }
  __syncthreads();
  #pragma unroll
  for (int nt = 0; nt < 4; nt++) {
    int col = nt*16 + (lane & 15), rb = (lane & 16) ? 8 : 0;
    float bias = ie2b[col] + pecb[col];
    #pragma unroll
    for (int v = 0; v < 8; v++)
      xt[wave][(rb+v)*64 + col] = (_Float16)(acc[nt][v] + bias);
  }
  __syncthreads();

  const float s_pm = rsqrtf(1.0f + 1e-3f);
  // pm1, pm2: 64->64, BN folded, relu, back through xt
  const _Float16* WT[2] = {pm1T, pm2T};
  const float* PB[2] = {pm1b, pm2b}; const float* PG[2] = {pm1g, pm2g};
  const float* PT[2] = {pm1bt, pm2bt};
  for (int L = 0; L < 2; L++) {
    v8f hh[4] = {};
    for (int kc = 0; kc < 64; kc += 32) {
      v16h a = ldA(xt[wave], 64, kc, lane);
      #pragma unroll
      for (int nt = 0; nt < 4; nt++)
        hh[nt] = WMMA(a, ldB(WT[L], 64, nt*16, kc, lane), hh[nt]);
    }
    __syncthreads();
    #pragma unroll
    for (int nt = 0; nt < 4; nt++) {
      int col = nt*16 + (lane & 15), rb = (lane & 16) ? 8 : 0;
      float s = PG[L][col]*s_pm, bt = PT[L][col], bs = PB[L][col];
      #pragma unroll
      for (int v = 0; v < 8; v++) {
        float x = (hh[nt][v] + bs)*s + bt;
        xt[wave][(rb+v)*64 + col] = (_Float16)fmaxf(x, 0.0f);
      }
    }
    __syncthreads();
  }
  // pm3: 64->32, BN, no relu
  v8f h3[2] = {};
  for (int kc = 0; kc < 64; kc += 32) {
    v16h a = ldA(xt[wave], 64, kc, lane);
    #pragma unroll
    for (int nt = 0; nt < 2; nt++)
      h3[nt] = WMMA(a, ldB(pm3T, 64, nt*16, kc, lane), h3[nt]);
  }
  __syncthreads();
  #pragma unroll
  for (int nt = 0; nt < 2; nt++) {
    int col = nt*16 + (lane & 15), rb = (lane & 16) ? 8 : 0;
    float s = pm3g[col]*s_pm, bt = pm3bt[col], bs = pm3b[col];
    #pragma unroll
    for (int v = 0; v < 8; v++)
      h3s[wave][(rb+v)*32 + col] = (_Float16)((h3[nt][v] + bs)*s + bt);
  }
  __syncthreads();
  // pm4: 32->1 dot, BN, softmax over the 16 neighbors of this (b,n)
  if (lane < 16) {
    float t = pm4b[0];
    #pragma unroll
    for (int k = 0; k < 32; k++) t += (float)h3s[wave][lane*32 + k] * pm4w[k];
    lg[wave][lane] = t * (pm4g[0]*s_pm) + pm4bt[0];
  }
  __syncthreads();
  if (lane < 16) {
    float m = lg[wave][0];
    #pragma unroll
    for (int k = 1; k < 16; k++) m = fmaxf(m, lg[wave][k]);
    float den = 0.0f;
    #pragma unroll
    for (int k = 0; k < 16; k++) den += expf(lg[wave][k] - m);
    attn[r0 + lane] = expf(lg[wave][lane] - m) / den;
  }
}

// ---- cost = attention-weighted sum over K, concat into [R2,128] -----------
__global__ void cost_kernel(const float* __restrict__ attn, const float* __restrict__ attnw,
                            const _Float16* __restrict__ f1, const _Float16* __restrict__ f3,
                            _Float16* __restrict__ cost) {
  int t = blockIdx.x * 256 + threadIdx.x;
  if (t >= R2 * 128) return;
  int bn = t >> 7, c = t & 127;
  const float*    a = (c < 64) ? attn : attnw;
  const _Float16* f = (c < 64) ? f1   : f3;
  int cc = c & 63;
  float s = 0.0f;
  #pragma unroll
  for (int k = 0; k < KNN; k++)
    s += a[bn*KNN + k] * (float)f[(size_t)(bn*KNN + k)*64 + cc];
  cost[(size_t)bn*128 + c] = (_Float16)s;
}

// ---------------------------------------------------------------------------
extern "C" void kernel_launch(void* const* d_in, const int* in_sizes, int n_in,
                              void* d_out, int out_size, void* d_ws, size_t ws_size,
                              hipStream_t stream) {
  const float* xyz1 = (const float*)d_in[0];
  const float* xyz2 = (const float*)d_in[1];
  const float* xyz2w = (const float*)d_in[2];
  const float* points1 = (const float*)d_in[3];
  const float* points2 = (const float*)d_in[4];
  const float* sf = (const float*)d_in[5];
  const float *cw1a=(const float*)d_in[6],  *cb1a=(const float*)d_in[7],
              *cw1b=(const float*)d_in[8],  *cb1b=(const float*)d_in[9],
              *cw2a=(const float*)d_in[10], *cb2a=(const float*)d_in[11],
              *cw2b=(const float*)d_in[12], *cb2b=(const float*)d_in[13],
              *cw3a=(const float*)d_in[14], *cb3a=(const float*)d_in[15],
              *cw3b=(const float*)d_in[16], *cb3b=(const float*)d_in[17],
              *cw4a=(const float*)d_in[18], *cb4a=(const float*)d_in[19],
              *cw4b=(const float*)d_in[20], *cb4b=(const float*)d_in[21];
  const float *ie_w1=(const float*)d_in[22], *ie_b1=(const float*)d_in[23],
              *ie_g =(const float*)d_in[24], *ie_bt=(const float*)d_in[25],
              *ie_w2=(const float*)d_in[26], *ie_b2=(const float*)d_in[27],
              *pec_w=(const float*)d_in[28], *pec_b=(const float*)d_in[29];
  const float *pm_w1=(const float*)d_in[30], *pm_b1=(const float*)d_in[31],
              *pm_g1=(const float*)d_in[32], *pm_bt1=(const float*)d_in[33],
              *pm_w2=(const float*)d_in[34], *pm_b2=(const float*)d_in[35],
              *pm_g2=(const float*)d_in[36], *pm_bt2=(const float*)d_in[37],
              *pm_w3=(const float*)d_in[38], *pm_b3=(const float*)d_in[39],
              *pm_g3=(const float*)d_in[40], *pm_bt3=(const float*)d_in[41],
              *pm_w4=(const float*)d_in[42], *pm_b4=(const float*)d_in[43],
              *pm_g4=(const float*)d_in[44], *pm_bt4=(const float*)d_in[45];

  char* w = (char*)d_ws;
  size_t off = 0;
  auto A = [&](size_t bytes) -> char* {
    char* p = w + off; off += (bytes + 255) & ~(size_t)255; return p;
  };
  int*      IDX   = (int*)A((size_t)RTOT*4);
  int*      IDXW  = (int*)A((size_t)RTOT*4);
  float*    DIRA  = (float*)A((size_t)RTOT*3*4);
  float*    DIRW  = (float*)A((size_t)RTOT*3*4);
  _Float16* ROWS  = (_Float16*)A((size_t)RTOT*128*2);
  _Float16* ROWS3 = (_Float16*)A((size_t)RTOT*96*2);
  _Float16* F1    = (_Float16*)A((size_t)RTOT*64*2);
  _Float16* F3    = (_Float16*)A((size_t)RTOT*64*2);
  float*    ATTN  = (float*)A((size_t)RTOT*4);
  float*    ATTNW = (float*)A((size_t)RTOT*4);
  _Float16* COST  = (_Float16*)A((size_t)R2*128*2);
  _Float16* W1aT = (_Float16*)A(64*128*2); _Float16* W1bT = (_Float16*)A(64*64*2);
  _Float16* W2aT = (_Float16*)A(64*128*2); _Float16* W2bT = (_Float16*)A(64*64*2);
  _Float16* W3aT = (_Float16*)A(64*96*2);  _Float16* W3bT = (_Float16*)A(64*64*2);
  _Float16* W4aT = (_Float16*)A(64*128*2); _Float16* W4bT = (_Float16*)A(64*64*2);
  _Float16* IE2T = (_Float16*)A(64*64*2);  _Float16* PECT = (_Float16*)A(64*32*2);
  _Float16* PM1T = (_Float16*)A(64*64*2);  _Float16* PM2T = (_Float16*)A(64*64*2);
  _Float16* PM3T = (_Float16*)A(32*64*2);

  auto prep = [&](const float* W_, _Float16* WT, int Cin, int Cout, int CinPad) {
    int total = Cout * CinPad;
    prep_wT<<<(total + 255)/256, 256, 0, stream>>>(W_, WT, Cin, Cout, CinPad);
  };
  prep(cw1a, W1aT, 128, 64, 128); prep(cw1b, W1bT, 64, 64, 64);
  prep(cw2a, W2aT, 128, 64, 128); prep(cw2b, W2bT, 64, 64, 64);
  prep(cw3a, W3aT,  67, 64,  96); prep(cw3b, W3bT, 64, 64, 64);
  prep(cw4a, W4aT, 128, 64, 128); prep(cw4b, W4bT, 64, 64, 64);
  prep(ie_w2, IE2T, 64, 64, 64);  prep(pec_w, PECT, 24, 64, 32);
  prep(pm_w1, PM1T, 64, 64, 64);  prep(pm_w2, PM2T, 64, 64, 64);
  prep(pm_w3, PM3T, 64, 32, 64);

  knn_kernel<<<dim3(NPTS/256, BATCH), 256, 0, stream>>>(xyz1, sf, xyz2, IDX);
  knn_kernel<<<dim3(NPTS/256, BATCH), 256, 0, stream>>>(xyz1, nullptr, xyz2w, IDXW);

  gather_dir<<<RTOT/256, 256, 0, stream>>>(xyz1, xyz2, IDX, IDXW, DIRA, DIRW, ROWS3);

  const int CHG = RTOT/16/WPB;   // 4096 blocks of 4 waves
  gather_feat<<<RTOT/256, 256, 0, stream>>>(points1, points2, IDX, ROWS);
  chain_kernel<<<CHG, 32*WPB, 0, stream>>>(ROWS, 128, W1aT, cb1a, W1bT, cb1b,
                                           F1, 64, 0, nullptr);
  gather_feat<<<RTOT/256, 256, 0, stream>>>(points1, points2, IDXW, ROWS);
  chain_kernel<<<CHG, 32*WPB, 0, stream>>>(ROWS, 128, W2aT, cb2a, W2bT, cb2b,
                                           ROWS3, 96, 3, nullptr);
  chain_kernel<<<CHG, 32*WPB, 0, stream>>>(ROWS3, 96, W3aT, cb3a, W3bT, cb3b,
                                           F3, 64, 0, nullptr);

  ipc_kernel<<<CHG, 32*WPB, 0, stream>>>(DIRA, ie_w1, ie_b1, ie_g, ie_bt,
      IE2T, ie_b2, PECT, pec_b, PM1T, pm_b1, pm_g1, pm_bt1,
      PM2T, pm_b2, pm_g2, pm_bt2, PM3T, pm_b3, pm_g3, pm_bt3,
      pm_w4, pm_b4, pm_g4, pm_bt4, ATTN);
  ipc_kernel<<<CHG, 32*WPB, 0, stream>>>(DIRW, ie_w1, ie_b1, ie_g, ie_bt,
      IE2T, ie_b2, PECT, pec_b, PM1T, pm_b1, pm_g1, pm_bt1,
      PM2T, pm_b2, pm_g2, pm_bt2, PM3T, pm_b3, pm_g3, pm_bt3,
      pm_w4, pm_b4, pm_g4, pm_bt4, ATTNW);

  cost_kernel<<<(R2*128)/256, 256, 0, stream>>>(ATTN, ATTNW, F1, F3, COST);

  chain_kernel<<<R2/16/WPB, 32*WPB, 0, stream>>>(COST, 128, W4aT, cb4a, W4bT, cb4b,
                                                 nullptr, 0, 0, (float*)d_out);
  (void)in_sizes; (void)n_in; (void)out_size; (void)ws_size;
}
